// TransformerGroundHead_91044716741010
// MI455X (gfx1250) — matrix operand order repaired and checked by reading, not compiled, not run on hardware
//
#include <hip/hip_runtime.h>
#include <hip/hip_bf16.h>

// ---------------------------------------------------------------------------
// Problem constants (from reference)
// ---------------------------------------------------------------------------
#define Bb 16
#define Tt 4096
#define Dd 768
#define N_BOX 1024
#define MAX_BBOX 128
#define DIM_DETR 256
#define TSPLIT 16

typedef __attribute__((ext_vector_type(16))) __bf16 v16bf;
typedef __attribute__((ext_vector_type(8)))  float  v8f;
// 4-byte-aligned float4 (features[:,1:] rows are only dword-aligned)
typedef float f32x4 __attribute__((ext_vector_type(4), aligned(4)));

// ---------------------------------------------------------------------------
// 1) Two-phase mean over T:  inputs (B,T,D) -> x (B,D)
//    192 threads x float4 = one full 768-float row per block iteration.
// ---------------------------------------------------------------------------
__global__ __launch_bounds__(192)
void mean_partial_kernel(const float* __restrict__ in, float* __restrict__ part) {
    const int d4 = threadIdx.x * 4;          // 0..764
    const int s  = blockIdx.y;               // 0..TSPLIT-1
    const int b  = blockIdx.z;               // 0..Bb-1
    const float* p = in + ((size_t)b * Tt + (size_t)s * (Tt / TSPLIT)) * Dd + d4;
    f32x4 sum = {0.f, 0.f, 0.f, 0.f};
    for (int t = 0; t < Tt / TSPLIT; ++t)
        sum += *(const f32x4*)(p + (size_t)t * Dd);
    *(f32x4*)(part + ((size_t)b * TSPLIT + s) * Dd + d4) = sum;
}

__global__ void mean_final_kernel(const float* __restrict__ part,
                                  float* __restrict__ x) {
    int i  = blockIdx.x * 256 + threadIdx.x;     // over Bb*Dd/4 = 3072
    int b  = i / (Dd / 4);
    int d4 = (i % (Dd / 4)) * 4;
    f32x4 s = {0.f, 0.f, 0.f, 0.f};
#pragma unroll
    for (int j = 0; j < TSPLIT; ++j)
        s += *(const f32x4*)(part + ((size_t)b * TSPLIT + j) * Dd + d4);
    *(f32x4*)(x + (size_t)b * Dd + d4) = s * (1.0f / (float)Tt);
}

// ---------------------------------------------------------------------------
// 2) WMMA GEMM:  Out[M,N] = act( A[M,K] @ W[N,K]^T + bias + Cin )
//    One wave per 16x16 tile; all flags compile-time; b128 operand loads.
// ---------------------------------------------------------------------------
template <int K, bool HAS_BIAS, bool HAS_CIN, bool RELU>
__global__ __launch_bounds__(32)
void gemm_wmma_kernel(const float* __restrict__ A, int lda,
                      const float* __restrict__ W, int ldw,
                      const float* __restrict__ bias,
                      const float* __restrict__ Cin,
                      float* __restrict__ Out, int ldo) {
    const int n0   = blockIdx.x * 16;
    const int m0   = blockIdx.y * 16;
    const int lane = threadIdx.x;          // 0..31
    const int half = lane >> 4;            // 0 or 1
    const int l    = lane & 15;

    v8f acc = {};
    if (HAS_CIN) {
        const float* crow = Cin + (size_t)(m0 + half * 8) * ldo + n0 + l;
#pragma unroll
        for (int r = 0; r < 8; ++r) acc[r] = crow[(size_t)r * ldo];
    }

    const float* arow = A + (size_t)(m0 + l) * lda;  // A row m0+l (K contiguous)
    const float* brow = W + (size_t)(n0 + l) * ldw;  // W row n0+l = B column n0+l

#pragma unroll 4
    for (int k0 = 0; k0 < K; k0 += 32) {
        // A 16x32 bf16 layout: lane half h holds K = k0+h*8+0..7 (VGPR0-3)
        //                      and           K = k0+16+h*8+0..7 (VGPR4-7)
        f32x4 a0 = *(const f32x4*)(arow + k0 + half * 8);
        f32x4 a1 = *(const f32x4*)(arow + k0 + half * 8 + 4);
        f32x4 a2 = *(const f32x4*)(arow + k0 + 16 + half * 8);
        f32x4 a3 = *(const f32x4*)(arow + k0 + 16 + half * 8 + 4);
        // B 32x16 bf16 layout: lane (half,l) holds column n0+l, K = k0+half*8+0..7 and k0+16+half*8+0..7
        f32x4 b0 = *(const f32x4*)(brow + k0 + half * 8);
        f32x4 b1 = *(const f32x4*)(brow + k0 + half * 8 + 4);
        f32x4 b2 = *(const f32x4*)(brow + k0 + 16 + half * 8);
        f32x4 b3 = *(const f32x4*)(brow + k0 + 16 + half * 8 + 4);

        v16bf a, b;
#pragma unroll
        for (int i = 0; i < 4; ++i) {
            a[i]      = (__bf16)a0[i];
            a[4 + i]  = (__bf16)a1[i];
            a[8 + i]  = (__bf16)a2[i];
            a[12 + i] = (__bf16)a3[i];
            b[i]      = (__bf16)b0[i];
            b[4 + i]  = (__bf16)b1[i];
            b[8 + i]  = (__bf16)b2[i];
            b[12 + i] = (__bf16)b3[i];
        }
        acc = __builtin_amdgcn_wmma_f32_16x16x32_bf16(
            /*neg_a=*/false, a, /*neg_b=*/false, b,
            /*c_mod=*/(short)0, acc, /*reuse_a=*/false, /*reuse_b=*/false);
    }

    const float bv = HAS_BIAS ? bias[n0 + l] : 0.0f;
    float* orow = Out + (size_t)(m0 + half * 8) * ldo + n0 + l;
    // C/D layout: VGPR r, lane half 0 -> M=r, half 1 -> M=8+r; N = l
#pragma unroll
    for (int r = 0; r < 8; ++r) {
        float v = acc[r] + bv;
        if (RELU) v = v > 0.0f ? v : 0.0f;
        orow[(size_t)r * ldo] = v;
    }
}

// ---------------------------------------------------------------------------
// 3) Gather x_boxes = xt[img]   (b128 copies, 192 threads = one row)
// ---------------------------------------------------------------------------
__global__ __launch_bounds__(192)
void gather_rows_kernel(const float* __restrict__ xt,
                        const int* __restrict__ bboxes,
                        float* __restrict__ xb) {
    const int box = blockIdx.x;                 // 0..N_BOX-1
    const int img = bboxes[box * 5];
    const int d4  = threadIdx.x * 4;
    *(f32x4*)(xb + (size_t)box * Dd + d4) =
        *(const f32x4*)(xt + (size_t)img * Dd + d4);
}

// ---------------------------------------------------------------------------
// 4) Segment counts / offsets / pos + att_mask (single block, 1024 threads)
// ---------------------------------------------------------------------------
__global__ __launch_bounds__(1024)
void segment_kernel(const int* __restrict__ bboxes,
                    int* __restrict__ pos,
                    float* __restrict__ att_mask) {
    __shared__ int sc[Bb];
    __shared__ int soff[Bb];
    int tid = threadIdx.x;
    if (tid < Bb) sc[tid] = 0;
    __syncthreads();
    int img = bboxes[tid * 5];
    atomicAdd(&sc[img], 1);
    __syncthreads();
    if (tid == 0) {
        int run = 0;
        for (int b = 0; b < Bb; ++b) { soff[b] = run; run += sc[b]; }
    }
    __syncthreads();
    pos[tid] = tid - soff[img];
    for (int i = tid; i < Bb * MAX_BBOX; i += 1024) {
        int b = i / MAX_BBOX, j = i % MAX_BBOX;
        att_mask[i] = (j < sc[b]) ? 1.0f : 0.0f;
    }
}

// ---------------------------------------------------------------------------
// 5) Zero vis_output region, then scatter h rows into it
// ---------------------------------------------------------------------------
__global__ void zero4_kernel(float* __restrict__ p, size_t n4) {
    size_t i = (size_t)blockIdx.x * 256 + threadIdx.x;
    if (i < n4) {
        f32x4 z = {0.f, 0.f, 0.f, 0.f};
        *(f32x4*)(p + i * 4) = z;
    }
}

__global__ __launch_bounds__(192)
void scatter_h_kernel(const float* __restrict__ h,
                      const int* __restrict__ bboxes,
                      const int* __restrict__ pos,
                      float* __restrict__ vis) {
    const int box = blockIdx.x;
    const int img = bboxes[box * 5];
    const int p   = pos[box];
    const int d4  = threadIdx.x * 4;
    *(f32x4*)(vis + ((size_t)img * MAX_BBOX + p) * Dd + d4) =
        *(const f32x4*)(h + (size_t)box * Dd + d4);
}

// ---------------------------------------------------------------------------
// Host-side launch
// ---------------------------------------------------------------------------
extern "C" void kernel_launch(void* const* d_in, const int* in_sizes, int n_in,
                              void* d_out, int out_size, void* d_ws, size_t ws_size,
                              hipStream_t stream) {
    const float* inputs   = (const float*)d_in[0];
    const int*   bboxes   = (const int*)  d_in[1];
    const float* features = (const float*)d_in[2];
    const float* t1w = (const float*)d_in[3];  const float* t1b = (const float*)d_in[4];
    const float* t2w = (const float*)d_in[5];  const float* t2b = (const float*)d_in[6];
    const float* d1w = (const float*)d_in[7];  const float* d1b = (const float*)d_in[8];
    const float* d2w = (const float*)d_in[9];  const float* d2b = (const float*)d_in[10];
    const float* m1w = (const float*)d_in[11]; const float* m1b = (const float*)d_in[12];
    const float* m2w = (const float*)d_in[13]; const float* m2b = (const float*)d_in[14];
    const float* pw  = (const float*)d_in[15]; const float* pb  = (const float*)d_in[16];

    float* out = (float*)d_out;
    float* vis      = out;                                    // Bb*MAX_BBOX*Dd
    float* att_mask = out + (size_t)Bb * MAX_BBOX * Dd;       // Bb*MAX_BBOX
    float* ret_x    = att_mask + (size_t)Bb * MAX_BBOX;       // Bb*Dd

    // Workspace layout (floats)
    float* ws = (float*)d_ws;
    float* part  = ws;                            // TSPLIT*Bb*Dd
    float* x     = part  + TSPLIT * Bb * Dd;      // Bb*Dd
    float* tmp16 = x     + Bb * Dd;               // Bb*Dd
    float* xt    = tmp16 + Bb * Dd;               // Bb*Dd
    float* fm1   = xt    + Bb * Dd;               // N_BOX*Dd
    float* fm    = fm1   + (size_t)N_BOX * Dd;    // N_BOX*Dd
    float* xb    = fm    + (size_t)N_BOX * Dd;    // N_BOX*Dd
    float* h     = xb    + (size_t)N_BOX * Dd;    // N_BOX*Dd
    int*   pos   = (int*)(h + (size_t)N_BOX * Dd);// N_BOX

    // ---- mean over T ----
    mean_partial_kernel<<<dim3(1, TSPLIT, Bb), 192, 0, stream>>>(inputs, part);
    mean_final_kernel<<<(Bb * Dd / 4 + 255) / 256, 256, 0, stream>>>(part, x);

    const dim3 wave(32);
    const dim3 grid_x16(Dd / 16, Bb / 16);        // M=16   -> (48, 1)
    const dim3 grid_box(Dd / 16, N_BOX / 16);     // M=1024 -> (48, 64)

    // ---- ret_x = relu(x@d1w.T+d1b) @ d2w.T + d2b ----
    gemm_wmma_kernel<Dd, true, false, true ><<<grid_x16, wave, 0, stream>>>(x, Dd, d1w, Dd, d1b, nullptr, tmp16, Dd);
    gemm_wmma_kernel<Dd, true, false, false><<<grid_x16, wave, 0, stream>>>(tmp16, Dd, d2w, Dd, d2b, nullptr, ret_x, Dd);

    // ---- xt = relu(x@t1w.T+t1b) @ t2w.T + t2b ----
    gemm_wmma_kernel<Dd, true, false, true ><<<grid_x16, wave, 0, stream>>>(x, Dd, t1w, Dd, t1b, nullptr, tmp16, Dd);
    gemm_wmma_kernel<Dd, true, false, false><<<grid_x16, wave, 0, stream>>>(tmp16, Dd, t2w, Dd, t2b, nullptr, xt, Dd);

    // ---- fm = relu(features[:,1:]@m1w.T+m1b) @ m2w.T + m2b ----
    gemm_wmma_kernel<DIM_DETR, true, false, true><<<grid_box, wave, 0, stream>>>(
        features + 1, 1 + DIM_DETR, m1w, DIM_DETR, m1b, nullptr, fm1, Dd);
    gemm_wmma_kernel<Dd, true, false, false><<<grid_box, wave, 0, stream>>>(fm1, Dd, m2w, Dd, m2b, nullptr, fm, Dd);

    // ---- x_boxes = xt[img] ----
    gather_rows_kernel<<<N_BOX, 192, 0, stream>>>(xt, bboxes, xb);

    // ---- h = concat([x_boxes, fm]) @ pw.T + pb  (pw split into two K=768 halves) ----
    gemm_wmma_kernel<Dd, true,  false, false><<<grid_box, wave, 0, stream>>>(xb, Dd, pw, 2 * Dd, pb, nullptr, h, Dd);
    gemm_wmma_kernel<Dd, false, true,  false><<<grid_box, wave, 0, stream>>>(fm, Dd, pw + Dd, 2 * Dd, nullptr, h, h, Dd);

    // ---- segment bookkeeping + att_mask ----
    segment_kernel<<<1, 1024, 0, stream>>>(bboxes, pos, att_mask);

    // ---- vis_output: zero then scatter ----
    const size_t vis_n4 = (size_t)Bb * MAX_BBOX * Dd / 4;
    zero4_kernel<<<(unsigned)((vis_n4 + 255) / 256), 256, 0, stream>>>(vis, vis_n4);
    scatter_h_kernel<<<N_BOX, 192, 0, stream>>>(h, bboxes, pos, vis);
}
